// LokrModule_16716012716545
// MI455X (gfx1250) — matrix operand order repaired and checked by reading, not compiled, not run on hardware
//
#include <hip/hip_runtime.h>
#include <hip/hip_bf16.h>

typedef __attribute__((ext_vector_type(16))) __bf16          v16bf;
typedef __attribute__((ext_vector_type(16))) unsigned short  u16x16;
typedef __attribute__((ext_vector_type(8)))  float           v8f;

#define M_DIM 8192
#define K_DIM 4096
#define N_DIM 4096
#define BM 256
#define BN 128
#define BK 32
#define BKP 36          // padded LDS row stride (halves): 18 words, cycle 32 -> conflict-free
#define NTHREADS 512    // 16 wave32
#define NC (K_DIM / BK) // 128 K-chunks
#define LORA_SCALE (1.0f / 16.0f)

// ---------------------------------------------------------------------------
// Cheap f32 -> (hi, lo) bf16 split via truncation (~3 VALU/elem).
// lo exactly compensates hi's truncation error up to lo's own 8-bit mantissa,
// so hi+lo carries ~16 mantissa bits (same as RNE split at 2x the VALU).
// ---------------------------------------------------------------------------
__device__ __forceinline__ void split_bf16_fast(float v, unsigned short& h, unsigned short& l) {
    const unsigned int u = __float_as_uint(v);
    h = (unsigned short)(u >> 16);
    const float r = v - __uint_as_float(u & 0xFFFF0000u);
    l = (unsigned short)(__float_as_uint(r) >> 16);
}

__device__ __forceinline__ void split_store4(const float4 v,
                                             unsigned short* __restrict__ hp,
                                             unsigned short* __restrict__ lp) {
    unsigned short h, l;
    split_bf16_fast(v.x, h, l); hp[0] = h; lp[0] = l;
    split_bf16_fast(v.y, h, l); hp[1] = h; lp[1] = l;
    split_bf16_fast(v.z, h, l); hp[2] = h; lp[2] = l;
    split_bf16_fast(v.w, h, l); hp[3] = h; lp[3] = l;
}

// ---------------------------------------------------------------------------
// Kernel 1: w2 = lokr_w2_a [64x16] @ lokr_w2_b [16x64]  -> 64x64 into d_ws
// ---------------------------------------------------------------------------
__global__ void lokr_w2_kernel(const float* __restrict__ w2a,
                               const float* __restrict__ w2b,
                               float* __restrict__ w2) {
    const int idx = blockIdx.x * blockDim.x + threadIdx.x;   // 0..4095
    const int r = idx >> 6;
    const int c = idx & 63;
    float s = 0.0f;
#pragma unroll
    for (int k = 0; k < 16; ++k)
        s += w2a[r * 16 + k] * w2b[k * 64 + c];
    w2[idx] = s;
}

// ---------------------------------------------------------------------------
// Kernel 2: bf16x3 split-precision WMMA GEMM, fused LoKr delta,
// double-buffered global->reg->convert->LDS pipeline.
// Block = 512 threads (16 wave32); block tile 256x128; wave tile 64x32.
// Per K=32 chunk per wave: 24 x v_wmma_f32_16x16x32_bf16.
// ---------------------------------------------------------------------------
__global__ __launch_bounds__(NTHREADS)
void lokr_gemm_kernel(const float* __restrict__ x,
                      const float* __restrict__ w,
                      const float* __restrict__ bias,
                      const float* __restrict__ w1,
                      const float* __restrict__ w2,
                      float* __restrict__ out) {
    __shared__ unsigned short AsH[2][BM * BKP];   // x tile, hi bf16 (double buffered)
    __shared__ unsigned short AsL[2][BM * BKP];   // x tile, lo bf16
    __shared__ unsigned short BsH[2][BN * BKP];   // (W+delta) tile, hi bf16
    __shared__ unsigned short BsL[2][BN * BKP];   // (W+delta) tile, lo bf16

    const int tid = threadIdx.x;
    const int m0  = blockIdx.y * BM;
    const int n0  = blockIdx.x * BN;

    const int wave  = tid >> 5;          // 0..15
    const int lane  = tid & 31;
    const int l     = lane & 15;
    const int half  = lane >> 4;
    const int waveM = (wave & 3) * 64;   // 4 waves along M (256)
    const int waveN = (wave >> 2) * 32;  // 4 waves along N (128)

    const int aK0 = half * 16;           // A frag elems 0..7  -> K = aK0 + e
    const int aK1 = half * 16 + 8;       // A frag elems 8..15 -> K = aK1 + (e-8)
    const int bK  = half * 16;           // B frag elems 0..15 -> K = bK + e

    v8f acc[4][2];
#pragma unroll
    for (int i = 0; i < 4; ++i)
#pragma unroll
        for (int j = 0; j < 2; ++j)
            acc[i][j] = (v8f){0.f, 0.f, 0.f, 0.f, 0.f, 0.f, 0.f, 0.f};

    // ---- loader lambdas -------------------------------------------------
    // A: 256 rows x 8 float4 = 2048 ids -> 4 per thread
    // B: 128 rows x 8 float4 = 1024 ids -> 2 per thread (delta fused in f32)
    float4 aReg[4], bReg[2];

    auto load_chunk = [&](int k0) {
#pragma unroll
        for (int i = 0; i < 4; ++i) {
            const int id  = tid + i * NTHREADS;
            const int row = id >> 3;
            const int c4  = (id & 7) << 2;
            aReg[i] = *reinterpret_cast<const float4*>(
                &x[(size_t)(m0 + row) * K_DIM + k0 + c4]);
        }
        const int kb = k0 >> 6;
        const int ko = k0 & 63;
#pragma unroll
        for (int i = 0; i < 2; ++i) {
            const int id  = tid + i * NTHREADS;
            const int row = id >> 3;
            const int c4  = (id & 7) << 2;
            const int n   = n0 + row;
            const float4 wv = *reinterpret_cast<const float4*>(
                &w[(size_t)n * K_DIM + k0 + c4]);
            const float  s1 = w1[(n >> 6) * 64 + kb] * LORA_SCALE;
            const float4 s2 = *reinterpret_cast<const float4*>(
                &w2[(n & 63) * 64 + ko + c4]);
            bReg[i].x = wv.x + s1 * s2.x;
            bReg[i].y = wv.y + s1 * s2.y;
            bReg[i].z = wv.z + s1 * s2.z;
            bReg[i].w = wv.w + s1 * s2.w;
        }
    };

    auto store_chunk = [&](int buf) {
#pragma unroll
        for (int i = 0; i < 4; ++i) {
            const int id  = tid + i * NTHREADS;
            const int row = id >> 3;
            const int c4  = (id & 7) << 2;
            split_store4(aReg[i], &AsH[buf][row * BKP + c4], &AsL[buf][row * BKP + c4]);
        }
#pragma unroll
        for (int i = 0; i < 2; ++i) {
            const int id  = tid + i * NTHREADS;
            const int row = id >> 3;
            const int c4  = (id & 7) << 2;
            split_store4(bReg[i], &BsH[buf][row * BKP + c4], &BsL[buf][row * BKP + c4]);
        }
    };

    // ---- prologue: stage chunk 0 into buffer 0 --------------------------
    load_chunk(0);
    store_chunk(0);
    __syncthreads();

    // ---- main pipeline ---------------------------------------------------
    for (int c = 0; c < NC; ++c) {
        const int cur = c & 1;

        // issue next chunk's global loads first (latency hides under WMMAs)
        if (c + 1 < NC) load_chunk((c + 1) * BK);

        // ---- fragments + 24 WMMAs on current buffer ----
        v16bf ah[4], bh[2];
#pragma unroll
        for (int i = 0; i < 4; ++i) {
            const unsigned short* p0 = &AsH[cur][(waveM + 16 * i + l) * BKP + aK0];
            const unsigned short* p1 = &AsH[cur][(waveM + 16 * i + l) * BKP + aK1];
            u16x16 t;
#pragma unroll
            for (int e = 0; e < 8; ++e) { t[e] = p0[e]; t[8 + e] = p1[e]; }
            ah[i] = __builtin_bit_cast(v16bf, t);
        }
#pragma unroll
        for (int j = 0; j < 2; ++j) {
            const unsigned short* p = &BsH[cur][(waveN + 16 * j + l) * BKP + bK];
            u16x16 t;
#pragma unroll
            for (int e = 0; e < 16; ++e) t[e] = p[e];
            bh[j] = __builtin_bit_cast(v16bf, t);
        }

        // term 1: hi * hi
#pragma unroll
        for (int i = 0; i < 4; ++i)
#pragma unroll
            for (int j = 0; j < 2; ++j)
                acc[i][j] = __builtin_amdgcn_wmma_f32_16x16x32_bf16(
                    false, ah[i], false, bh[j], (short)0, acc[i][j], false, false);

        // term 2: hi(A) * lo(B)
        {
            v16bf bl[2];
#pragma unroll
            for (int j = 0; j < 2; ++j) {
                const unsigned short* p = &BsL[cur][(waveN + 16 * j + l) * BKP + bK];
                u16x16 t;
#pragma unroll
                for (int e = 0; e < 16; ++e) t[e] = p[e];
                bl[j] = __builtin_bit_cast(v16bf, t);
            }
#pragma unroll
            for (int i = 0; i < 4; ++i)
#pragma unroll
                for (int j = 0; j < 2; ++j)
                    acc[i][j] = __builtin_amdgcn_wmma_f32_16x16x32_bf16(
                        false, ah[i], false, bl[j], (short)0, acc[i][j], false, false);
        }

        // term 3: lo(A) * hi(B)
        {
            v16bf al[4];
#pragma unroll
            for (int i = 0; i < 4; ++i) {
                const unsigned short* p0 = &AsL[cur][(waveM + 16 * i + l) * BKP + aK0];
                const unsigned short* p1 = &AsL[cur][(waveM + 16 * i + l) * BKP + aK1];
                u16x16 t;
#pragma unroll
                for (int e = 0; e < 8; ++e) { t[e] = p0[e]; t[8 + e] = p1[e]; }
                al[i] = __builtin_bit_cast(v16bf, t);
            }
#pragma unroll
            for (int i = 0; i < 4; ++i)
#pragma unroll
                for (int j = 0; j < 2; ++j)
                    acc[i][j] = __builtin_amdgcn_wmma_f32_16x16x32_bf16(
                        false, al[i], false, bh[j], (short)0, acc[i][j], false, false);
        }

        // convert + store next chunk into the other buffer (WMMAs co-execute)
        if (c + 1 < NC) store_chunk(cur ^ 1);

        __syncthreads();
    }

    // ---- epilogue: add bias, write 64x32 patch per wave ----
#pragma unroll
    for (int i = 0; i < 4; ++i) {
#pragma unroll
        for (int j = 0; j < 2; ++j) {
            const int col = n0 + waveN + 16 * j + l;
            const float bv = bias[col];
#pragma unroll
            for (int r = 0; r < 8; ++r) {
                const int row = m0 + waveM + 16 * i + r + 8 * half;
                out[(size_t)row * N_DIM + col] = acc[i][j][r] + bv;
            }
        }
    }
}

// ---------------------------------------------------------------------------
extern "C" void kernel_launch(void* const* d_in, const int* in_sizes, int n_in,
                              void* d_out, int out_size, void* d_ws, size_t ws_size,
                              hipStream_t stream) {
    const float* x    = (const float*)d_in[0];   // [4,2048,4096]
    const float* w    = (const float*)d_in[1];   // [4096,4096]
    const float* bias = (const float*)d_in[2];   // [4096]
    const float* w1   = (const float*)d_in[3];   // [64,64]
    const float* w2a  = (const float*)d_in[4];   // [64,16]
    const float* w2b  = (const float*)d_in[5];   // [16,64]
    float* out = (float*)d_out;                  // [4,2048,4096]
    float* w2  = (float*)d_ws;                   // [64,64] scratch

    lokr_w2_kernel<<<16, 256, 0, stream>>>(w2a, w2b, w2);

    dim3 grid(N_DIM / BN, M_DIM / BM);           // (32, 32)
    lokr_gemm_kernel<<<grid, NTHREADS, 0, stream>>>(x, w, bias, w1, w2, out);
}
